// WindowAttention_644245094961
// MI455X (gfx1250) — matrix operand order repaired and checked by reading, not compiled, not run on hardware
//
#include <hip/hip_runtime.h>

typedef _Float16 h16;
typedef __attribute__((ext_vector_type(16))) _Float16 v16h;
typedef __attribute__((ext_vector_type(8)))  float    v8f;

#define N_WIN 80
#define C_DIM 256

// ---- LDS layout (bytes) ----
#define SM_QKVS 0        // 80 x 768 f16 = 122880
#define SM_XS   122880   // 80 x 256 f16 = 40960  (x during QKV, ys during attn/proj)
#define SM_TBL  163840   // 285 x 8 f32  = 9120
#define SM_MSK  172960   // 80 x 80 f32  = 25600
#define SM_SBUF 198560   // 80 x 84 f32  = 26880
#define SM_PBUF 225440   // 80 x 96 f16  = 15360
#define SM_WPAN 240800   // 64 x 256 f16 = 32768
#define SM_IDX  273568   // 80 x 80 u16  = 12800 (rel-pos index * 8)
#define SM_TOTAL 286368

// ---- workspace layout (bytes) ----
#define WS_TBL  0        // 285*8 f32
#define WS_WQH  9216     // 768*256 f16
#define WS_WPH  402432   // 256*256 f16
#define WS_QKVB 533504   // 768 f32 fused qkv bias

__device__ __forceinline__ v8f wmma16(v16h a, v16h b, v8f c) {
  return __builtin_amdgcn_wmma_f32_16x16x32_f16(false, a, false, b, (short)0, c,
                                                false, false);
}

// A tile 16x32 f16, rows row0..row0+15, K k0..k0+31, from row-major [m][k] LDS
__device__ __forceinline__ v16h load_a16(const h16* lds, int row0, int k0,
                                         int stride, int lane) {
  int m  = lane & 15;
  int kb = (lane & 16) ? 16 : 0;
  const h16* p = lds + (row0 + m) * stride + k0;
  v16h a;
#pragma unroll
  for (int i = 0; i < 16; ++i) a[i] = p[kb + i];
  return a;
}

// B tile 32x16, B[k][n] = src[n][k] (src row-major [n][k])
__device__ __forceinline__ v16h load_bT(const h16* lds, int col0, int k0,
                                        int stride, int lane) {
  int n  = lane & 15;
  int kb = (lane & 16) ? 16 : 0;
  const h16* p = lds + (col0 + n) * stride + k0 + kb;
  v16h b;
#pragma unroll
  for (int i = 0; i < 16; ++i) b[i] = p[i];
  return b;
}

// B tile 32x16, B[k][n] = src[k][n] (src row-major [k][n]); zero for k >= kvalid
__device__ __forceinline__ v16h load_bD(const h16* lds, int k0, int col0,
                                        int stride, int kvalid, int lane) {
  int n  = lane & 15;
  int kb = (lane & 16) ? 16 : 0;
  v16h b;
#pragma unroll
  for (int i = 0; i < 16; ++i) {
    int k = k0 + kb + i;
    b[i] = (k < kvalid) ? lds[k * stride + col0 + n] : (h16)0.f;
  }
  return b;
}

// ---------------- prep kernels ----------------

__global__ void cpb_mlp_kernel(const float* __restrict__ w1,
                               const float* __restrict__ b1,
                               const float* __restrict__ w2,
                               float* __restrict__ tbl_out) {
  int e = blockIdx.x * blockDim.x + threadIdx.x;
  if (e >= 285) return;                      // (2*8-1)*(2*10-1)
  int i = e / 19, j = e % 19;
  float fy = (float)(i - 7) / 7.0f * 8.0f;
  float fx = (float)(j - 9) / 9.0f * 8.0f;
  const float inv_log2_8 = 1.0f / 3.0f;      // 1/log2(8)
  float ty = copysignf(log2f(fabsf(fy) + 1.0f), fy) * inv_log2_8;
  float tx = copysignf(log2f(fabsf(fx) + 1.0f), fx) * inv_log2_8;
  float acc[8];
#pragma unroll
  for (int h = 0; h < 8; ++h) acc[h] = 0.f;
  for (int r = 0; r < 512; ++r) {
    float hv = fmaxf(w1[r * 2 + 0] * ty + w1[r * 2 + 1] * tx + b1[r], 0.f);
#pragma unroll
    for (int h = 0; h < 8; ++h) acc[h] += w2[h * 512 + r] * hv;
  }
#pragma unroll
  for (int h = 0; h < 8; ++h)
    tbl_out[e * 8 + h] = 16.0f / (1.0f + expf(-acc[h]));
}

__global__ void convert_w_kernel(const float* __restrict__ wq,
                                 const float* __restrict__ wp,
                                 const float* __restrict__ q_bias,
                                 const float* __restrict__ v_bias,
                                 h16* __restrict__ wqh, h16* __restrict__ wph,
                                 float* __restrict__ qkvb) {
  int i = blockIdx.x * blockDim.x + threadIdx.x;
  if (i < 768 * 256) wqh[i] = (h16)wq[i];
  if (i < 256 * 256) wph[i] = (h16)wp[i];
  if (i < 768)
    qkvb[i] = (i < 256) ? q_bias[i] : (i >= 512 ? v_bias[i - 512] : 0.f);
}

// ---------------- fused window-attention kernel ----------------

__launch_bounds__(256, 1)
__global__ void win_attn_kernel(const float* __restrict__ x,
                                const float* __restrict__ mask,
                                const float* __restrict__ logit_scale,
                                const float* __restrict__ tbl,
                                const h16*  __restrict__ wqh,
                                const h16*  __restrict__ wph,
                                const float* __restrict__ qkvb,
                                const float* __restrict__ proj_b,
                                float* __restrict__ out) {
  extern __shared__ char smem[];
  h16*   qkvs = (h16*)(smem + SM_QKVS);   // [80][768]
  h16*   xs   = (h16*)(smem + SM_XS);     // [80][256] x, later ys
  float* tblb = (float*)(smem + SM_TBL);  // [285][8]
  float* msk  = (float*)(smem + SM_MSK);  // [80][80]
  float* sbuf = (float*)(smem + SM_SBUF); // [80][84]
  h16*   pbuf = (h16*)(smem + SM_PBUF);   // [80][96]
  h16*   wpan = (h16*)(smem + SM_WPAN);   // [64][256]
  unsigned short* idxm = (unsigned short*)(smem + SM_IDX); // [80][80], *8

  const int b    = blockIdx.x;
  const int tid  = threadIdx.x;
  const int wave = tid >> 5;
  const int lane = tid & 31;

  // ---- stage x (f32->f16), bias table, mask window, rel-pos index map ----
  const float* xb = x + (size_t)b * (N_WIN * C_DIM);
  for (int i = tid; i < N_WIN * C_DIM; i += 256) xs[i] = (h16)xb[i];
  for (int i = tid; i < 285 * 8; i += 256) tblb[i] = tbl[i];
  const float* mw = mask + (size_t)(b & 63) * (N_WIN * N_WIN);
  for (int i = tid; i < N_WIN * N_WIN; i += 256) msk[i] = mw[i];
  for (int i = tid; i < N_WIN * N_WIN; i += 256) {
    int m = i / 80, n2 = i % 80;
    int qy = m / 10, qx = m % 10, ky = n2 / 10, kx = n2 % 10;
    idxm[i] = (unsigned short)((((qy - ky + 7) * 19) + (qx - kx + 9)) * 8);
  }
  __syncthreads();

  // ---- QKV GEMM: (80x256) @ Wqkv^T -> qkvs (80x768), 64-col weight panels ----
  for (int pc = 0; pc < 768; pc += 64) {
    const uint4* src4 = (const uint4*)(wqh + (size_t)pc * 256);
    uint4* dst4 = (uint4*)wpan;
    for (int i = tid; i < (64 * 256) / 8; i += 256) dst4[i] = src4[i];
    if (pc + 64 < 768)  // prefetch next panel into cache while computing
      __builtin_prefetch((const char*)(wqh + (size_t)(pc + 64) * 256) + tid * 128,
                         0, 3);
    __syncthreads();
    for (int t = wave; t < 20; t += 8) {      // 4 col-tiles x 5 row-tiles
      int nt = t / 5, mt = t % 5;
      int col0 = nt * 16, row0 = mt * 16;
      v8f c = {};
      for (int k0 = 0; k0 < 256; k0 += 32) {
        v16h a  = load_a16(xs, row0, k0, 256, lane);
        v16h bm = load_bT(wpan, col0, k0, 256, lane);
        c = wmma16(a, bm, c);
      }
      int col = pc + col0 + (lane & 15);
      float bias = qkvb[col];
      int mbase = row0 + ((lane & 16) ? 8 : 0);
#pragma unroll
      for (int r = 0; r < 8; ++r)
        qkvs[(mbase + r) * 768 + col] = (h16)(c[r] + bias);
    }
    __syncthreads();
  }

  // ---- normalize q,k rows; fold exp(min(logit_scale,log100)) into q ----
  for (int t = tid; t < 1280; t += 256) {     // 80 rows * 8 heads * {q,k}
    int part = t & 1;
    int h    = (t >> 1) & 7;
    int n    = t >> 4;
    h16* p = qkvs + n * 768 + part * 256 + h * 32;
    float ss = 0.f;
#pragma unroll
    for (int d = 0; d < 32; ++d) { float v = (float)p[d]; ss += v * v; }
    float inv = 1.0f / (sqrtf(ss) + 5e-5f);
    if (part == 0)
      inv *= expf(fminf(logit_scale[h], 4.6051701859880914f)); // log(100)
#pragma unroll
    for (int d = 0; d < 32; ++d) p[d] = (h16)((float)p[d] * inv);
  }
  __syncthreads();

  // ---- attention, heads sequential, all 8 waves cooperate ----
  for (int h = 0; h < 8; ++h) {
    // S = qn @ kn^T  (+bias +mask) -> sbuf
    for (int t = wave; t < 25; t += 8) {
      int mt = t / 5, nt = t % 5;
      int row0 = mt * 16, col0 = nt * 16;
      v16h a  = load_a16(qkvs, row0, h * 32, 768, lane);
      v16h bm = load_bT(qkvs + 256 + h * 32, col0, 0, 768, lane);
      v8f c = {};
      c = wmma16(a, bm, c);
      int ncol  = col0 + (lane & 15);
      int mbase = row0 + ((lane & 16) ? 8 : 0);
#pragma unroll
      for (int r = 0; r < 8; ++r) {
        int m = mbase + r;
        sbuf[m * 84 + ncol] =
            c[r] + tblb[idxm[m * 80 + ncol] + h] + msk[m * 80 + ncol];
      }
    }
    __syncthreads();
    // row softmax -> pbuf (f16, zero-padded to K=96)
    for (int n = tid; n < 80; n += 256) {
      float mx = -1e30f;
      for (int j = 0; j < 80; ++j) mx = fmaxf(mx, sbuf[n * 84 + j]);
      float sum = 0.f;
      for (int j = 0; j < 80; ++j) {
        float e = expf(sbuf[n * 84 + j] - mx);
        sum += e;
        sbuf[n * 84 + j] = e;
      }
      float inv = 1.0f / sum;
      for (int j = 0; j < 80; ++j) pbuf[n * 96 + j] = (h16)(sbuf[n * 84 + j] * inv);
      for (int j = 80; j < 96; ++j) pbuf[n * 96 + j] = (h16)0.f;
    }
    __syncthreads();
    // out_h = P(80x96,padded) @ V(80x32) -> ys (= xs region)
    for (int t = wave; t < 10; t += 8) {      // 5 row-tiles x 2 col-tiles
      int mt = t / 2, dt = t % 2;
      int row0 = mt * 16, col0 = dt * 16;
      v8f c = {};
      for (int k0 = 0; k0 < 96; k0 += 32) {
        v16h a  = load_a16(pbuf, row0, k0, 96, lane);
        v16h bm = load_bD(qkvs + 512 + h * 32, k0, col0, 768, 80, lane);
        c = wmma16(a, bm, c);
      }
      int ncol  = col0 + (lane & 15);
      int mbase = row0 + ((lane & 16) ? 8 : 0);
#pragma unroll
      for (int r = 0; r < 8; ++r)
        xs[(mbase + r) * 256 + h * 32 + ncol] = (h16)c[r];
    }
    __syncthreads();
  }

  // ---- projection: ys(80x256) @ Wp^T + b -> out (f32) ----
  float* ob = out + (size_t)b * (N_WIN * C_DIM);
  for (int pc = 0; pc < 256; pc += 64) {
    const uint4* src4 = (const uint4*)(wph + (size_t)pc * 256);
    uint4* dst4 = (uint4*)wpan;
    for (int i = tid; i < (64 * 256) / 8; i += 256) dst4[i] = src4[i];
    if (pc + 64 < 256)
      __builtin_prefetch((const char*)(wph + (size_t)(pc + 64) * 256) + tid * 128,
                         0, 3);
    __syncthreads();
    for (int t = wave; t < 20; t += 8) {
      int nt = t / 5, mt = t % 5;
      int col0 = nt * 16, row0 = mt * 16;
      v8f c = {};
      for (int k0 = 0; k0 < 256; k0 += 32) {
        v16h a  = load_a16(xs, row0, k0, 256, lane);
        v16h bm = load_bT(wpan, col0, k0, 256, lane);
        c = wmma16(a, bm, c);
      }
      int ncol  = pc + col0 + (lane & 15);
      int mbase = row0 + ((lane & 16) ? 8 : 0);
      float pb = proj_b[ncol];
#pragma unroll
      for (int r = 0; r < 8; ++r)
        ob[(mbase + r) * 256 + ncol] = c[r] + pb;
    }
    __syncthreads();
  }
}

// ---------------- launcher ----------------

extern "C" void kernel_launch(void* const* d_in, const int* in_sizes, int n_in,
                              void* d_out, int out_size, void* d_ws, size_t ws_size,
                              hipStream_t stream) {
  const float* x           = (const float*)d_in[0];
  const float* mask        = (const float*)d_in[1];
  const float* w_qkv       = (const float*)d_in[2];
  const float* q_bias      = (const float*)d_in[3];
  const float* v_bias      = (const float*)d_in[4];
  const float* logit_scale = (const float*)d_in[5];
  const float* cpb_w1      = (const float*)d_in[6];
  const float* cpb_b1      = (const float*)d_in[7];
  const float* cpb_w2      = (const float*)d_in[8];
  const float* proj_w      = (const float*)d_in[9];
  const float* proj_b      = (const float*)d_in[10];
  float* out = (float*)d_out;

  char* ws = (char*)d_ws;
  float* tbl  = (float*)(ws + WS_TBL);
  h16*   wqh  = (h16*)(ws + WS_WQH);
  h16*   wph  = (h16*)(ws + WS_WPH);
  float* qkvb = (float*)(ws + WS_QKVB);

  cpb_mlp_kernel<<<5, 64, 0, stream>>>(cpb_w1, cpb_b1, cpb_w2, tbl);
  convert_w_kernel<<<(768 * 256 + 255) / 256, 256, 0, stream>>>(
      w_qkv, proj_w, q_bias, v_bias, wqh, wph, qkvb);

  (void)hipFuncSetAttribute((const void*)win_attn_kernel,
                            hipFuncAttributeMaxDynamicSharedMemorySize,
                            SM_TOTAL);
  win_attn_kernel<<<2048, 256, SM_TOTAL, stream>>>(
      x, mask, logit_scale, tbl, wqh, wph, qkvb, proj_b, out);
}
